// MultiQueryAttention_21423296873307
// MI455X (gfx1250) — matrix operand order repaired and checked by reading, not compiled, not run on hardware
//
#include <hip/hip_runtime.h>
#include <hip/hip_bf16.h>

typedef __attribute__((ext_vector_type(16))) __bf16 v16bf;
typedef __attribute__((ext_vector_type(8)))  float  v8f;
typedef __attribute__((ext_vector_type(4)))  unsigned int v4u;
typedef __attribute__((ext_vector_type(8)))  int  v8i;
typedef __attribute__((ext_vector_type(4)))  int  v4i;

#define BATCH 2
#define SEQ   2048
#define DIN   1024
#define DOUT  1024
#define NHEAD 16
#define HDIM  64
#define BT    (BATCH*SEQ)

// ---- float -> bf16 (round to nearest even) ----
__device__ __forceinline__ __bf16 f2bf(float f) {
    union { float f; unsigned u; } in; in.f = f;
    unsigned u = in.u;
    u += 0x7FFFu + ((u >> 16) & 1u);
    unsigned short hs = (unsigned short)(u >> 16);
    union { unsigned short s; __bf16 b; } out; out.s = hs;
    return out.b;
}

// Assemble a 16-element bf16 fragment from two 16-byte chunks (two b128 loads).
union frag_u { v4u q[2]; v16bf v; };
__device__ __forceinline__ v16bf load_frag2(const __bf16* p0, const __bf16* p1) {
    frag_u u;
    u.q[0] = *(const v4u*)p0;
    u.q[1] = *(const v4u*)p1;
    return u.v;
}
__device__ __forceinline__ v16bf load_a_frag(const __bf16* row_k0, int laneHi) {
    const __bf16* p = row_k0 + 16 * laneHi;
    return load_frag2(p, p + 8);
}
__device__ __forceinline__ v16bf load_b_frag(const __bf16* p) {
    return load_frag2(p, p + 8);
}

// ---- TDM: 2-D tensor tile -> LDS via Tensor Data Mover (D# per CDNA5 ISA Ch.8) ----
// data_size = 2 bytes (bf16). tile (tile_d0 x tile_d1) from a row-major tensor whose
// rows are stride0 elements apart; gaddr points at the tile start. tensor_d0/d1 bound
// the remaining tensor for OOB handling. lds_off is a byte offset in the wave's LDS.
__device__ __forceinline__ void tdm_load_2d_bf16(const void* gaddr, unsigned lds_off,
                                                 unsigned tensor_d0, unsigned tensor_d1,
                                                 unsigned tile_d0, unsigned tile_d1,
                                                 unsigned stride0) {
    unsigned long long ga = (unsigned long long)gaddr;
    v4u g0;
    g0[0] = 1u;                                              // count=1 (valid user D#)
    g0[1] = lds_off;                                         // lds_addr (bytes)
    g0[2] = (unsigned)(ga & 0xFFFFFFFFu);                    // global_addr[31:0]
    g0[3] = (unsigned)((ga >> 32) & 0x01FFFFFFu) | (2u << 30); // addr[56:32] | type=2
    v8i g1;
    g1[0] = (int)(1u << 16);                                 // data_size=1 -> 2B; mask=0
    g1[1] = (int)((tensor_d0 & 0xFFFFu) << 16);              // tensor_dim0[15:0]
    g1[2] = (int)(((tensor_d0 >> 16) & 0xFFFFu) | ((tensor_d1 & 0xFFFFu) << 16));
    g1[3] = (int)(((tensor_d1 >> 16) & 0xFFFFu) | ((tile_d0 & 0xFFFFu) << 16));
    g1[4] = (int)(tile_d1 & 0xFFFFu);                        // tile_dim1; tile_dim2=0
    g1[5] = (int)stride0;                                    // tensor_dim0_stride[31:0]
    g1[6] = 0;                                               // stride0 hi | stride1 lo
    g1[7] = 0;
    v4i z4 = {0, 0, 0, 0};
    v8i z8 = {0, 0, 0, 0, 0, 0, 0, 0};
    // 6-arg form (clang-23 / therock-10.0 headers): (g0, g1, g2, g3, g4, cpol)
    __builtin_amdgcn_tensor_load_to_lds(g0, g1, z4, z4, z8, 0);
}
__device__ __forceinline__ unsigned lds_offset_of(const void* shared_ptr) {
    // generic address of an LDS object: low 32 bits are the LDS byte offset
    return (unsigned)(unsigned long long)shared_ptr;
}

// ======================= converters / packers =======================
__global__ void k_convert_x(const float* __restrict__ x, __bf16* __restrict__ xb, int n) {
    int i = blockIdx.x * blockDim.x + threadIdx.x;
    if (i < n) xb[i] = f2bf(x[i]);
}

// W [N][K] fp32 -> Bp [(K/32)][N][32] bf16 (fragment-native packed layout for WMMA B)
__global__ void k_pack_w(const float* __restrict__ W, __bf16* __restrict__ Bp, int N, int K) {
    int i = blockIdx.x * blockDim.x + threadIdx.x;
    if (i >= N * K) return;
    int n = i / K, k = i % K;
    Bp[((size_t)(k >> 5) * N + n) * 32 + (k & 31)] = f2bf(W[i]);
}

// ======================= generic WMMA GEMM =======================
// C[M][N] fp32 = A[M][K] (bf16 row-major) * Bp (packed [(K/32)][N][32] bf16)
// grid = (M/32, N/64), block = 32: one wave owns a 32x64 tile (8 WMMAs / K-step).
__global__ void k_gemm_bf16(const __bf16* __restrict__ A, const __bf16* __restrict__ Bp,
                            float* __restrict__ C, int M, int N, int K) {
    const int mt = blockIdx.x, nt = blockIdx.y;
    const int lane = threadIdx.x & 31;
    const int laneLo = lane & 15, laneHi = lane >> 4;

    const __bf16* arow0 = A + (size_t)(mt * 32 + laneLo) * K;
    const __bf16* arow1 = arow0 + (size_t)16 * K;
    v8f acc[2][4] = {};

    for (int k0 = 0; k0 < K; k0 += 32) {
        if (k0 + 64 < K) __builtin_prefetch(arow0 + k0 + 64, 0, 1);  // global_prefetch_b8
        v16bf af0 = load_a_frag(arow0 + k0, laneHi);
        v16bf af1 = load_a_frag(arow1 + k0, laneHi);
        const __bf16* bbase = Bp + ((size_t)(k0 >> 5) * N + nt * 64 + laneLo) * 32 + 16 * laneHi;
        #pragma unroll
        for (int nf = 0; nf < 4; ++nf) {
            v16bf bfr = load_b_frag(bbase + (size_t)nf * 16 * 32);
            acc[0][nf] = __builtin_amdgcn_wmma_f32_16x16x32_bf16(
                false, af0, false, bfr, (short)0, acc[0][nf], false, false);
            acc[1][nf] = __builtin_amdgcn_wmma_f32_16x16x32_bf16(
                false, af1, false, bfr, (short)0, acc[1][nf], false, false);
        }
    }
    #pragma unroll
    for (int mh = 0; mh < 2; ++mh)
        #pragma unroll
        for (int nf = 0; nf < 4; ++nf)
            #pragma unroll
            for (int r = 0; r < 8; ++r)
                C[(size_t)(mt * 32 + mh * 16 + r + 8 * laneHi) * N + nt * 64 + nf * 16 + laneLo]
                    = acc[mh][nf][r];
}

// ======================= RoPE epilogues =======================
__global__ void k_rope_q(const float* __restrict__ qr, const float* __restrict__ cosb,
                         const float* __restrict__ sinb, __bf16* __restrict__ qb) {
    size_t idx = (size_t)blockIdx.x * blockDim.x + threadIdx.x;
    if (idx >= (size_t)BT * NHEAD * 32) return;
    int d = idx & 31; size_t rem = idx >> 5;
    int h = rem % NHEAD; rem /= NHEAD;
    int t = rem % SEQ;   int b = rem / SEQ;
    const float* row = qr + ((size_t)(b * SEQ + t)) * DOUT + h * HDIM;
    float q1 = row[d], q2 = row[d + 32];
    float o1 = q1 * cosb[t * HDIM + d]      - q2 * sinb[t * HDIM + d];
    float o2 = q2 * cosb[t * HDIM + d + 32] + q1 * sinb[t * HDIM + d + 32];
    __bf16* out = qb + (((size_t)(b * NHEAD + h) * SEQ) + t) * HDIM;
    out[d] = f2bf(o1);
    out[d + 32] = f2bf(o2);
}

__global__ void k_rope_k(const float* __restrict__ kr, const float* __restrict__ cosb,
                         const float* __restrict__ sinb, __bf16* __restrict__ kb) {
    size_t idx = (size_t)blockIdx.x * blockDim.x + threadIdx.x;
    if (idx >= (size_t)BT * 32) return;
    int d = idx & 31; size_t bt = idx >> 5;
    int t = bt % SEQ;
    const float* row = kr + bt * HDIM;
    float k1 = row[d], k2 = row[d + 32];
    float o1 = k1 * cosb[t * HDIM + d]      - k2 * sinb[t * HDIM + d];
    float o2 = k2 * cosb[t * HDIM + d + 32] + k1 * sinb[t * HDIM + d + 32];
    kb[bt * HDIM + d]      = f2bf(o1);
    kb[bt * HDIM + d + 32] = f2bf(o2);
}

// v_raw [B][T][64] fp32 -> vT [B][64][T] bf16 (so P*V B-fragments are contiguous over s)
__global__ void k_transpose_v(const float* __restrict__ vr, __bf16* __restrict__ vt) {
    size_t idx = (size_t)blockIdx.x * blockDim.x + threadIdx.x;
    if (idx >= (size_t)BT * HDIM) return;
    int d = idx & 63; size_t bt = idx >> 6;
    int t = bt % SEQ; int b = bt / SEQ;
    vt[((size_t)b * HDIM + d) * SEQ + t] = f2bf(vr[idx]);
}

// ======================= flash attention (TDM-fed, double-buffered) =======================
// grid = B*H*(T/16) blocks of 32 threads; one wave per 16-query tile of one (b,h).
// K/V tiles are DMA'd into LDS by the Tensor Data Mover while the wave computes.
__global__ void k_attn(const __bf16* __restrict__ qb, const __bf16* __restrict__ kb,
                       const __bf16* __restrict__ vt, __bf16* __restrict__ ctxb) {
    const int QT = SEQ / 16;
    int tile = blockIdx.x;
    int qt = tile % QT;
    int bh = tile / QT;
    int h = bh % NHEAD, b = bh / NHEAD;
    const int lane = threadIdx.x & 31;
    const int laneLo = lane & 15, laneHi = lane >> 4;

    const __bf16* qhead  = qb + (size_t)(b * NHEAD + h) * SEQ * HDIM;
    const __bf16* kbatch = kb + (size_t)b * SEQ * HDIM;
    const __bf16* vtb    = vt + (size_t)b * HDIM * SEQ;

    __shared__ __bf16 ldsK[2][32 * HDIM];   // K tile: [s 0..31][d 0..63]
    __shared__ __bf16 ldsV[2][HDIM * 32];   // V^T tile: [d 0..63][s 0..31]
    __shared__ __bf16 pTile[16 * 32];

    const unsigned kOff[2] = { lds_offset_of(&ldsK[0][0]), lds_offset_of(&ldsK[1][0]) };
    const unsigned vOff[2] = { lds_offset_of(&ldsV[0][0]), lds_offset_of(&ldsV[1][0]) };

    // persistent Q A-fragments over d=[0,32) and [32,64)
    const __bf16* qrow = qhead + (size_t)(qt * 16 + laneLo) * HDIM;
    v16bf aq[2];
    aq[0] = load_a_frag(qrow, laneHi);
    aq[1] = load_a_frag(qrow + 32, laneHi);

    v8f o[4] = {};
    float m[8], l[8];
    #pragma unroll
    for (int r = 0; r < 8; ++r) { m[r] = -__builtin_inff(); l[r] = 0.0f; }

    const float scale = 0.125f;     // 1/sqrt(64)
    const int sEnd = qt * 16 + 16;  // causal bound

    // kick off TDM for the first K/V tile (s0 = 0) into buffer 0
    tdm_load_2d_bf16(kbatch, kOff[0], HDIM, SEQ, HDIM, 32, HDIM);
    tdm_load_2d_bf16(vtb, vOff[0], SEQ, HDIM, 32, HDIM, SEQ);

    for (int s0 = 0; s0 < sEnd; s0 += 32) {
        const int buf = (s0 >> 5) & 1;
        // prefetch next tile into the other buffer while this one is consumed
        if (s0 + 32 < sEnd) {
            const int nxt = buf ^ 1;
            tdm_load_2d_bf16(kbatch + (size_t)(s0 + 32) * HDIM, kOff[nxt],
                             HDIM, SEQ - (s0 + 32), HDIM, 32, HDIM);
            tdm_load_2d_bf16(vtb + (s0 + 32), vOff[nxt],
                             SEQ - (s0 + 32), HDIM, 32, HDIM, SEQ);
            __builtin_amdgcn_s_wait_tensorcnt(2);  // current tile done; next in flight
        } else {
            __builtin_amdgcn_s_wait_tensorcnt(0);
        }
        const __bf16* lk = &ldsK[buf][0];
        const __bf16* lv = &ldsV[buf][0];

        // ---- S = Q * K^T : two 16x16 C frags; K^T frags via ds_load_b128 ----
        v8f sc[2] = {};
        #pragma unroll
        for (int nf = 0; nf < 2; ++nf) {
            const __bf16* krow = lk + (size_t)(nf * 16 + laneLo) * HDIM + 16 * laneHi;
            #pragma unroll
            for (int kk = 0; kk < 2; ++kk) {
                v16bf bk = load_b_frag(krow + kk * 32);
                sc[nf] = __builtin_amdgcn_wmma_f32_16x16x32_bf16(
                    false, aq[kk], false, bk, (short)0, sc[nf], false, false);
            }
        }

        // ---- scale + causal mask ----
        #pragma unroll
        for (int nf = 0; nf < 2; ++nf)
            #pragma unroll
            for (int r = 0; r < 8; ++r) {
                int trow = qt * 16 + r + 8 * laneHi;
                int scol = s0 + nf * 16 + laneLo;
                float val = sc[nf][r] * scale;
                sc[nf][r] = (scol > trow) ? -__builtin_inff() : val;
            }

        // ---- online softmax (row reductions within 16-lane halves) ----
        #pragma unroll
        for (int r = 0; r < 8; ++r) {
            float v = fmaxf(sc[0][r], sc[1][r]);
            #pragma unroll
            for (int mk = 1; mk < 16; mk <<= 1) v = fmaxf(v, __shfl_xor(v, mk, 32));
            float mn = fmaxf(m[r], v);
            float corr = __expf(m[r] - mn);
            m[r] = mn;
            float p0 = __expf(sc[0][r] - mn);
            float p1 = __expf(sc[1][r] - mn);
            sc[0][r] = p0; sc[1][r] = p1;
            float rs = p0 + p1;
            #pragma unroll
            for (int mk = 1; mk < 16; mk <<= 1) rs += __shfl_xor(rs, mk, 32);
            l[r] = l[r] * corr + rs;
            #pragma unroll
            for (int nf = 0; nf < 4; ++nf) o[nf][r] *= corr;
        }

        // ---- P (C-layout fp32) -> LDS bf16 16x32 row-major ----
        #pragma unroll
        for (int nf = 0; nf < 2; ++nf)
            #pragma unroll
            for (int r = 0; r < 8; ++r)
                pTile[(r + 8 * laneHi) * 32 + nf * 16 + laneLo] = f2bf(sc[nf][r]);
        __syncthreads();

        // ---- reload P as A-fragment: two ds_load_b128 ----
        v16bf ap = load_a_frag(&pTile[laneLo * 32], laneHi);

        // ---- O += P * V : V^T tile rows contiguous over s -> ds_load_b128 ----
        #pragma unroll
        for (int nf = 0; nf < 4; ++nf) {
            const __bf16* vrow = lv + (size_t)(nf * 16 + laneLo) * 32 + 16 * laneHi;
            v16bf bv = load_b_frag(vrow);
            o[nf] = __builtin_amdgcn_wmma_f32_16x16x32_bf16(
                false, ap, false, bv, (short)0, o[nf], false, false);
        }
        __syncthreads();
    }

    // ---- normalize and write ctx (bf16) as [B*T][H*64] ----
    #pragma unroll
    for (int nf = 0; nf < 4; ++nf)
        #pragma unroll
        for (int r = 0; r < 8; ++r) {
            int t = qt * 16 + r + 8 * laneHi;
            float val = o[nf][r] / l[r];
            ctxb[(size_t)(b * SEQ + t) * DOUT + h * HDIM + nf * 16 + laneLo] = f2bf(val);
        }
}

// ======================= launcher =======================
extern "C" void kernel_launch(void* const* d_in, const int* in_sizes, int n_in,
                              void* d_out, int out_size, void* d_ws, size_t ws_size,
                              hipStream_t stream) {
    const float* x    = (const float*)d_in[0];
    // d_in[1] = mask (causal, handled analytically)
    const float* cosb = (const float*)d_in[2];
    const float* sinb = (const float*)d_in[3];
    const float* Wq   = (const float*)d_in[4];
    const float* Wk   = (const float*)d_in[5];
    const float* Wv   = (const float*)d_in[6];
    const float* Wo   = (const float*)d_in[7];
    float* out = (float*)d_out;

    char* p = (char*)d_ws;
    auto alloc = [&](size_t bytes) -> void* {
        void* r = (void*)p;
        p += (bytes + 255) & ~(size_t)255;
        return r;
    };
    __bf16* xb    = (__bf16*)alloc((size_t)BT * DIN * 2);
    __bf16* WqP   = (__bf16*)alloc((size_t)DIN * DOUT * 2);
    __bf16* WkP   = (__bf16*)alloc((size_t)DIN * HDIM * 2);
    __bf16* WvP   = (__bf16*)alloc((size_t)DIN * HDIM * 2);
    __bf16* WoP   = (__bf16*)alloc((size_t)DOUT * DIN * 2);
    float*  q_raw = (float*)alloc((size_t)BT * DOUT * 4);
    float*  k_raw = (float*)alloc((size_t)BT * HDIM * 4);
    float*  v_raw = (float*)alloc((size_t)BT * HDIM * 4);
    __bf16* qb    = (__bf16*)alloc((size_t)BT * DOUT * 2);   // [B][H][T][64]
    __bf16* kbuf  = (__bf16*)alloc((size_t)BT * HDIM * 2);   // [B][T][64]
    __bf16* vtb   = (__bf16*)alloc((size_t)BT * HDIM * 2);   // [B][64][T]
    __bf16* ctxb  = (__bf16*)alloc((size_t)BT * DOUT * 2);
    (void)ws_size; (void)in_sizes; (void)n_in; (void)out_size;

    const int TB = 256;

    // 1) bf16 conversion + fragment-native weight packing
    {
        int n = BT * DIN;
        k_convert_x<<<(n + TB - 1) / TB, TB, 0, stream>>>(x, xb, n);
        int nw = DOUT * DIN;
        k_pack_w<<<(nw + TB - 1) / TB, TB, 0, stream>>>(Wq, WqP, DOUT, DIN);
        int nk = HDIM * DIN;
        k_pack_w<<<(nk + TB - 1) / TB, TB, 0, stream>>>(Wk, WkP, HDIM, DIN);
        k_pack_w<<<(nk + TB - 1) / TB, TB, 0, stream>>>(Wv, WvP, HDIM, DIN);
        k_pack_w<<<(nw + TB - 1) / TB, TB, 0, stream>>>(Wo, WoP, DIN, DOUT);
    }

    // 2) projections
    k_gemm_bf16<<<dim3(BT / 32, DOUT / 64), 32, 0, stream>>>(xb, WqP, q_raw, BT, DOUT, DIN);
    k_gemm_bf16<<<dim3(BT / 32, HDIM / 64), 32, 0, stream>>>(xb, WkP, k_raw, BT, HDIM, DIN);
    k_gemm_bf16<<<dim3(BT / 32, HDIM / 64), 32, 0, stream>>>(xb, WvP, v_raw, BT, HDIM, DIN);

    // 3) RoPE + repack
    {
        size_t nq = (size_t)BT * NHEAD * 32;
        k_rope_q<<<(unsigned)((nq + TB - 1) / TB), TB, 0, stream>>>(q_raw, cosb, sinb, qb);
        size_t nk = (size_t)BT * 32;
        k_rope_k<<<(unsigned)((nk + TB - 1) / TB), TB, 0, stream>>>(k_raw, cosb, sinb, kbuf);
        size_t nv = (size_t)BT * HDIM;
        k_transpose_v<<<(unsigned)((nv + TB - 1) / TB), TB, 0, stream>>>(v_raw, vtb);
    }

    // 4) flash attention (causal, streaming softmax, TDM-fed K/V tiles)
    k_attn<<<BATCH * NHEAD * (SEQ / 16), 32, 0, stream>>>(qb, kbuf, vtb, ctxb);

    // 5) output projection -> d_out fp32
    k_gemm_bf16<<<dim3(BT / 32, DIN / 64), 32, 0, stream>>>(ctxb, WoP, out, BT, DIN, DOUT);
}
